// ResonanceAttention_25374666785529
// MI455X (gfx1250) — compile-verified, matrix operand order checked by reading, not run on hardware
//
#include <hip/hip_runtime.h>

// ResonanceAttention for MI455X (gfx1250, wave32, WMMA bf16 w/ f32 accumulate).
//
// Shapes: B=2, L=2048, D=1024, H=16, DH=64.
// d_out = [ final (B*L*D f32) | attn (B*H*L*L f32) ].
// d_ws  = [ qkv bf16 4096x3072 | phase f32 4096x16 | O f32 4096x1024 ] ~40.3MB.
// qkv is kept in bf16: it is only ever consumed as WMMA operands, the whole
// tensor (24MB) lives in the 192MB L2, and K/Q tiles become raw async copies.

typedef __bf16 bf16;
typedef __attribute__((ext_vector_type(16))) __bf16 v16bf;
typedef __attribute__((ext_vector_type(8)))  float  v8f;
typedef __attribute__((ext_vector_type(4)))  int    v4i;

#define AS1 __attribute__((address_space(1)))
#define AS3 __attribute__((address_space(3)))

#if defined(__has_builtin)
#if __has_builtin(__builtin_amdgcn_global_load_async_to_lds_b128)
#define HAVE_ASYNC_LDS 1
#endif
#endif
#ifndef HAVE_ASYNC_LDS
#define HAVE_ASYNC_LDS 0
#endif

// 16-byte global->LDS copy; async (ASYNCcnt-tracked) when the builtin exists.
// Signature confirmed by probe: (v4i AS1*, v4i AS3*, imm offset, imm cpol).
__device__ __forceinline__ void copy16_g2l(const void* g, void* l) {
#if HAVE_ASYNC_LDS
  __builtin_amdgcn_global_load_async_to_lds_b128(
      (AS1 v4i*)const_cast<void*>(g), (AS3 v4i*)l, 0, 0);
#else
  *(uint4*)l = *(const uint4*)g;
#endif
}

__device__ __forceinline__ void wait_async0() {
#if HAVE_ASYNC_LDS
#if __has_builtin(__builtin_amdgcn_s_wait_asynccnt)
  __builtin_amdgcn_s_wait_asynccnt(0);
#else
  asm volatile("s_wait_asynccnt 0x0" ::: "memory");
#endif
#endif
}

// ---------------------------------------------------------------------------
// WMMA 16x16x32 bf16 fragment load from an LDS tile (ISA 7.12.2 layouts).
// A: lane l<16 = row M=l; lane-half selects K-halves {8h..8h+7, 16+8h..}.
// Row-major [16 x >=K] tile -> two 16B chunks per lane.  B fragments use the
// identical pattern when the B tile is staged transposed ([N][K]).
__device__ __forceinline__ v16bf lds_frag(const bf16* base, int rowStrideElems,
                                          int kOff) {
  const int lane = threadIdx.x & 31;
  const int r    = lane & 15;
  const int half = lane >> 4;
  const char* p = (const char*)base +
                  (size_t)r * rowStrideElems * sizeof(bf16) +
                  (size_t)kOff * sizeof(bf16) + half * 16;
  union { v16bf v; struct { uint4 a; uint4 b; } q; } u;
  u.q.a = *(const uint4*)(p);
  u.q.b = *(const uint4*)(p + 32);
  return u.v;
}

// ---------------------------------------------------------------------------
// C[M,N] = A[M,K] @ B[K,N] + bias.  f32 inputs, bf16 WMMA, OutT output.
// N, K compile-time -> strength-reduced addressing.  Register double-buffered
// staging: next tile's global loads issue before the current tile's WMMAs.
template <int N, int K, typename OutT>
__global__ __launch_bounds__(256) void gemm_bf16_kernel(
    const float* __restrict__ A, const float* __restrict__ Bm,
    const float* __restrict__ bias, OutT* __restrict__ C) {
  __shared__ bf16 As[128 * 32];   // row-major [m][k]
  __shared__ bf16 Bts[64 * 32];   // transposed [n][k]

  const int tid  = threadIdx.x;
  const int w    = tid >> 5;
  const int lane = tid & 31;
  const int r    = lane & 15;
  const int half = lane >> 4;
  const int m0   = blockIdx.y * 128;
  const int n0   = blockIdx.x * 64;

  const int arow = tid >> 5;    // A stage: rows arow+8i, col acol
  const int acol = tid & 31;
  const int brow = tid >> 6;    // B stage: k rows brow+4i, col bcol
  const int bcol = tid & 63;
  const bool bok = (n0 + bcol) < N;

  const float* aptr = A + (size_t)(m0 + arow) * K + acol;
  const float* bptr = Bm + (size_t)brow * N + n0 + bcol;

  float aReg[16], bReg[8];
  auto loadAB = [&](int k0) {
#pragma unroll
    for (int i = 0; i < 16; ++i) aReg[i] = aptr[(size_t)8 * i * K + k0];
#pragma unroll
    for (int i = 0; i < 8; ++i)
      bReg[i] = bok ? bptr[(size_t)(k0 + 4 * i) * N] : 0.0f;
  };

  v8f acc[4];
#pragma unroll
  for (int c = 0; c < 4; ++c) acc[c] = {};

  loadAB(0);
  for (int k0 = 0; k0 < K; k0 += 32) {
#pragma unroll
    for (int i = 0; i < 16; ++i) As[(arow + 8 * i) * 32 + acol] = (bf16)aReg[i];
#pragma unroll
    for (int i = 0; i < 8; ++i) Bts[bcol * 32 + brow + 4 * i] = (bf16)bReg[i];
    __syncthreads();
    if (k0 + 32 < K) loadAB(k0 + 32);   // overlap with WMMA below

    v16bf a  = lds_frag(As + w * 16 * 32, 32, 0);
    v16bf b0 = lds_frag(Bts + 0 * 16 * 32, 32, 0);
    v16bf b1 = lds_frag(Bts + 1 * 16 * 32, 32, 0);
    v16bf b2 = lds_frag(Bts + 2 * 16 * 32, 32, 0);
    v16bf b3 = lds_frag(Bts + 3 * 16 * 32, 32, 0);
    acc[0] = __builtin_amdgcn_wmma_f32_16x16x32_bf16(false, a, false, b0, (short)0, acc[0], false, false);
    acc[1] = __builtin_amdgcn_wmma_f32_16x16x32_bf16(false, a, false, b1, (short)0, acc[1], false, false);
    acc[2] = __builtin_amdgcn_wmma_f32_16x16x32_bf16(false, a, false, b2, (short)0, acc[2], false, false);
    acc[3] = __builtin_amdgcn_wmma_f32_16x16x32_bf16(false, a, false, b3, (short)0, acc[3], false, false);
    __syncthreads();
  }

  // C/D layout: VGPR v holds M = v + 8*half, N = (lane&15).
#pragma unroll
  for (int c = 0; c < 4; ++c) {
    int col = n0 + c * 16 + r;
    if (col < N) {
      float bia = bias[col];
#pragma unroll
      for (int v = 0; v < 8; ++v) {
        int row = m0 + w * 16 + v + 8 * half;
        C[(size_t)row * N + col] = (OutT)(acc[c][v] + bia);
      }
    }
  }
}

// ---------------------------------------------------------------------------
// Fused attention: per workgroup = 16 query rows of one (b,h).
// Full 16x2048 score strip lives in LDS (320KB WGP LDS); the 536MB attn
// tensor is written to HBM exactly once.  K tiles double-buffer via
// GLOBAL_LOAD_ASYNC_TO_LDS (raw bf16, no VGPR round trip).
//
// Dynamic LDS map (bytes):
//   Sf  f32 [16][2048]  @      0 .. 131072   (head reused as f32 O-accum)
//   Pb  bf16[16][2048]  @ 131072 .. 196608
//   Qb  bf16[16][64]    @ 196608 .. 198656
//   KV0 bf16 16KB       @ 198656 .. 215040   (K tile [128][64] / V^T [64][128])
//   KV1 bf16 16KB       @ 215040 .. 231424
//   phj f32 [2048]      @ 231424 .. 239616
//   phi f32 [16]        @ 239616 .. 239680
#define ATTN_LDS_BYTES 239680

__global__ __launch_bounds__(256) void resonance_attn_kernel(
    const bf16* __restrict__ qkv,     // [B*L, 3072] bf16: [q|k|v] blocks
    const float* __restrict__ phase,  // [B*L, 16]
    float* __restrict__ attn_out,     // [B,H,L,L]
    float* __restrict__ o_out) {      // [B*L, 1024]
  constexpr int L = 2048;
  extern __shared__ char smem[];
  float* Sf  = (float*)(smem);
  bf16*  Pb  = (bf16*)(smem + 131072);
  bf16*  Qb  = (bf16*)(smem + 196608);
  bf16*  KV0 = (bf16*)(smem + 198656);
  bf16*  KV1 = (bf16*)(smem + 215040);
  float* phj = (float*)(smem + 231424);
  float* phi = (float*)(smem + 239616);

  const int tid  = threadIdx.x;
  const int w    = tid >> 5;
  const int lane = tid & 31;
  const int r    = lane & 15;
  const int half = lane >> 4;
  const int l0   = blockIdx.x * 16;
  const int b    = blockIdx.y >> 4;
  const int h    = blockIdx.y & 15;
  const int bL   = b * L;

  // 128-row K-tile stage: raw 16B async copies (1024 chunks / 256 threads).
  auto stage_k = [&](bf16* dst, int j0) {
#pragma unroll
    for (int i = 0; i < 4; ++i) {
      int e = tid + 256 * i;
      int jl = e >> 3, dh0 = (e & 7) * 8;
      copy16_g2l(qkv + (size_t)(bL + j0 + jl) * 3072 + 1024 + h * 64 + dh0,
                 dst + jl * 64 + dh0);
    }
  };

  // --- Stage Q (raw bf16; 1/8 scale folded into score epilogue), phases ---
  if (tid < 128) {
    int row = tid >> 3, dh0 = (tid & 7) * 8;
    copy16_g2l(qkv + (size_t)(bL + l0 + row) * 3072 + h * 64 + dh0,
               Qb + row * 64 + dh0);
  }
  if (tid < 16) phi[tid] = phase[(size_t)(bL + l0 + tid) * 16 + h];
#pragma unroll
  for (int i = 0; i < 8; ++i) {
    int e = tid + 256 * i;
    phj[e] = phase[(size_t)(bL + e) * 16 + h];
  }
  stage_k(KV0, 0);                     // prefetch first K tile
  wait_async0();
  __syncthreads();

  // Loop-invariant values the compiler can't hoist past LDS barriers itself:
  // the two Q fragments and this thread's 8 query-phase values.
  const v16bf aQ0 = lds_frag(Qb, 64, 0);
  const v16bf aQ1 = lds_frag(Qb, 64, 32);
  float phm[8];
#pragma unroll
  for (int v = 0; v < 8; ++v) phm[v] = phi[v + 8 * half];

  // --- Scores: S[16][2048] = (Q K^T)/8 - (pi - pj)^2 ----------------------
  for (int jt = 0; jt < 16; ++jt) {
    const int j0 = jt * 128;
    bf16* cur = (jt & 1) ? KV1 : KV0;
    if (jt + 1 < 16) stage_k((jt & 1) ? KV0 : KV1, j0 + 128);  // async ahead

    const int c0 = w * 16;             // wave's 16 key columns of this chunk
    v8f acc = {};
    {
      v16bf bk0 = lds_frag(cur + c0 * 64, 64, 0);
      v16bf bk1 = lds_frag(cur + c0 * 64, 64, 32);
      acc = __builtin_amdgcn_wmma_f32_16x16x32_bf16(
          false, aQ0, false, bk0, (short)0, acc, false, false);
      acc = __builtin_amdgcn_wmma_f32_16x16x32_bf16(
          false, aQ1, false, bk1, (short)0, acc, false, false);
    }
    const int col = j0 + c0 + r;
    const float pj = phj[col];
#pragma unroll
    for (int v = 0; v < 8; ++v) {
      int m = v + 8 * half;
      float dph = phm[v] - pj;
      Sf[m * 2048 + col] = acc[v] * 0.125f - dph * dph;
    }
    wait_async0();
    __syncthreads();
  }

  // --- Softmax per row; wave w owns rows {w, w+8}; write attn once --------
#pragma unroll
  for (int rr = 0; rr < 2; ++rr) {
    const int m = w + rr * 8;
    float* Srow = Sf + m * 2048;
    float mx = -3.4e38f;
    for (int c = lane; c < 2048; c += 32) mx = fmaxf(mx, Srow[c]);
    for (int off = 16; off > 0; off >>= 1) mx = fmaxf(mx, __shfl_xor(mx, off, 32));
    float sum = 0.0f;
    for (int c = lane; c < 2048; c += 32) {
      float e = __expf(Srow[c] - mx);
      Srow[c] = e;
      sum += e;
    }
    for (int off = 16; off > 0; off >>= 1) sum += __shfl_xor(sum, off, 32);
    const float inv = 1.0f / sum;
    float* arow = attn_out + (((size_t)(b * 16 + h) * 2048) + (l0 + m)) * 2048;
    bf16* prow = Pb + m * 2048;
    for (int c = lane; c < 2048; c += 32) {
      float p = Srow[c] * inv;
      arow[c] = p;                // the single HBM write of the attn tensor
      prow[c] = (bf16)p;          // bf16 copy for P@V below
    }
  }
  __syncthreads();

  // --- O[16][64] = P[16][2048] @ V[2048][64] ------------------------------
  float* Oacc = Sf;                    // reuse score region as f32 accum
  for (int i = tid; i < 16 * 64; i += 256) Oacc[i] = 0.0f;

  bf16* Vt = KV0;                      // V^T staged [dh][jl], stride 128
  const int kstep = w >> 1;
  const int chalf = w & 1;
  v8f oacc[2];
  oacc[0] = {}; oacc[1] = {};
  for (int jt = 0; jt < 16; ++jt) {
    const int j0 = jt * 128;
    __syncthreads();                   // prior Vt consumers done
#pragma unroll
    for (int i = 0; i < 32; ++i) {
      int e = tid + 256 * i;
      int jl = e >> 6, dh = e & 63;    // coalesced bf16 reads along dh
      Vt[dh * 128 + jl] = qkv[(size_t)(bL + j0 + jl) * 3072 + 2048 + h * 64 + dh];
    }
    __syncthreads();
    v16bf a = lds_frag(Pb, 2048, j0 + kstep * 32);
#pragma unroll
    for (int s = 0; s < 2; ++s) {
      int n0 = chalf * 32 + s * 16;
      v16bf bv = lds_frag(Vt + n0 * 128, 128, kstep * 32);
      oacc[s] = __builtin_amdgcn_wmma_f32_16x16x32_bf16(
          false, a, false, bv, (short)0, oacc[s], false, false);
    }
  }
  __syncthreads();
#pragma unroll
  for (int s = 0; s < 2; ++s) {
#pragma unroll
    for (int v = 0; v < 8; ++v) {
      int m = v + 8 * half;
      int col = chalf * 32 + s * 16 + r;
      atomicAdd(&Oacc[m * 64 + col], oacc[s][v]);
    }
  }
  __syncthreads();
#pragma unroll
  for (int i = 0; i < 4; ++i) {
    int e = tid + 256 * i;
    int row = e >> 6, col = e & 63;
    o_out[(size_t)(bL + l0 + row) * 1024 + h * 64 + col] = Oacc[row * 64 + col];
  }
}

// ---------------------------------------------------------------------------
extern "C" void kernel_launch(void* const* d_in, const int* in_sizes, int n_in,
                              void* d_out, int out_size, void* d_ws, size_t ws_size,
                              hipStream_t stream) {
  (void)in_sizes; (void)n_in; (void)out_size; (void)ws_size;
  const float* x       = (const float*)d_in[0];  // [2,2048,1024]
  const float* W_qkv   = (const float*)d_in[1];  // [1024,3072]
  const float* b_qkv   = (const float*)d_in[2];  // [3072]
  const float* W_phase = (const float*)d_in[3];  // [1024,16]
  const float* b_phase = (const float*)d_in[4];  // [16]
  const float* W_out   = (const float*)d_in[5];  // [1024,1024]
  const float* b_out   = (const float*)d_in[6];  // [1024]

  float* out  = (float*)d_out;                          // [2,2048,1024]
  float* attn = out + (size_t)2 * 2048 * 1024;          // [2,16,2048,2048]

  bf16*  qkvb = (bf16*)d_ws;                            // [4096,3072] bf16
  float* phw  = (float*)(qkvb + (size_t)4096 * 3072);   // [4096,16] f32
  float* ow   = phw + (size_t)4096 * 16;                // [4096,1024] f32

  // 1) qkv = x @ W_qkv + b_qkv  -> bf16 workspace (WMMA-operand precision)
  gemm_bf16_kernel<3072, 1024, bf16><<<dim3(3072 / 64, 4096 / 128), 256, 0, stream>>>(
      x, W_qkv, b_qkv, qkvb);
  // 2) phase = x @ W_phase + b_phase  (N=16, zero-padded B tile) -> f32
  gemm_bf16_kernel<16, 1024, float><<<dim3(1, 4096 / 128), 256, 0, stream>>>(
      x, W_phase, b_phase, phw);
  // 3) fused scores + resonance bias + softmax + attn write + P@V
  (void)hipFuncSetAttribute((const void*)resonance_attn_kernel,
                            hipFuncAttributeMaxDynamicSharedMemorySize,
                            ATTN_LDS_BYTES);
  resonance_attn_kernel<<<dim3(2048 / 16, 2 * 16), 256, ATTN_LDS_BYTES, stream>>>(
      qkvb, phw, attn, ow);
  // 4) final = O @ W_out + b_out
  gemm_bf16_kernel<1024, 1024, float><<<dim3(1024 / 64, 4096 / 128), 256, 0, stream>>>(
      ow, W_out, b_out, out);
}
